// TopKMoE_19980187861739
// MI455X (gfx1250) — matrix operand order repaired and checked by reading, not compiled, not run on hardware
//
#include <hip/hip_runtime.h>
#include <hip/hip_bf16.h>

typedef __attribute__((ext_vector_type(16))) __bf16 v16bf;
typedef __attribute__((ext_vector_type(8)))  float  v8f;

#define NB   128      // tokens (B)
#define SS   64       // seq len
#define DD   2048     // model dim
#define NE   8        // experts
#define HH   8192     // hidden
#define OO   10       // out dim
#define HT   128      // h-columns per block
#define KT   32       // K per WMMA step (bf16)
#define NSUB (HT / 16)  // 8 N-subtiles per block

union FragBF { v16bf v; unsigned u[8]; };
union FragC  { v8f  v; float f[8]; };

// fp32 pair -> packed bf16x2, round-to-nearest (+0x8000) + v_perm_b32 pack
__device__ __forceinline__ unsigned pack2(float lo, float hi) {
    unsigned ul = __float_as_uint(lo) + 0x8000u;
    unsigned uh = __float_as_uint(hi) + 0x8000u;
    return __builtin_amdgcn_perm(uh, ul, 0x07060302u);  // {uh[31:16], ul[31:16]}
}

// ---------------------------------------------------------------------------
// Kernel 1: gating (softmax -> top2 -> renorm) + zero ypart + x -> bf16 pack
// ---------------------------------------------------------------------------
__global__ __launch_bounds__(256) void moe_gate_kernel(
    const float* __restrict__ E, const float* __restrict__ gw,
    const float* __restrict__ gb, float* __restrict__ wout,
    float* __restrict__ ypart, unsigned* __restrict__ xbf)
{
    __shared__ float red[NE * 256];
    const int b = blockIdx.x;
    const int t = threadIdx.x;

    if (t < NE * OO) ypart[b * NE * OO + t] = 0.0f;

    const float* x = E + (size_t)b * SS * DD + (size_t)(SS - 1) * DD;

    // pack this token's x row to bf16 pairs (1024 dwords per row)
#pragma unroll
    for (int i = 0; i < 4; ++i) {
        int p = i * 256 + t;                       // pair index 0..1023
        float2 v = *(const float2*)(x + p * 2);
        xbf[(size_t)b * (DD / 2) + p] = pack2(v.x, v.y);
    }

    float l[NE];
#pragma unroll
    for (int e = 0; e < NE; ++e) l[e] = 0.0f;
    for (int d = t; d < DD; d += 256) {
        float xv = x[d];
#pragma unroll
        for (int e = 0; e < NE; ++e) l[e] += xv * gw[e * DD + d];
    }
#pragma unroll
    for (int e = 0; e < NE; ++e) red[e * 256 + t] = l[e];
    __syncthreads();
    for (int s = 128; s > 0; s >>= 1) {
        if (t < s) {
#pragma unroll
            for (int e = 0; e < NE; ++e)
                red[e * 256 + t] += red[e * 256 + t + s];
        }
        __syncthreads();
    }
    if (t == 0) {
        float lg[NE], p[NE];
        float mx = -1e30f;
#pragma unroll
        for (int e = 0; e < NE; ++e) { lg[e] = red[e * 256] + gb[e]; mx = fmaxf(mx, lg[e]); }
        float s = 0.0f;
#pragma unroll
        for (int e = 0; e < NE; ++e) { p[e] = __expf(lg[e] - mx); s += p[e]; }
#pragma unroll
        for (int e = 0; e < NE; ++e) p[e] /= s;
        int i1 = 0;
#pragma unroll
        for (int e = 1; e < NE; ++e) if (p[e] > p[i1]) i1 = e;
        int i2 = (i1 == 0) ? 1 : 0;
#pragma unroll
        for (int e = 0; e < NE; ++e) if (e != i1 && p[e] > p[i2]) i2 = e;
        float denom = p[i1] + p[i2] + 1e-9f;
#pragma unroll
        for (int e = 0; e < NE; ++e) wout[b * NE + e] = 0.0f;
        wout[b * NE + i1] = p[i1] / denom;
        wout[b * NE + i2] = p[i2] / denom;
    }
}

// ---------------------------------------------------------------------------
// Kernel 2: fused FFN.  grid = (H/HT, NE), block = 256 (8 wave32).
//   A fragments: direct global_load_b128 from L2-resident bf16 x
//   B fragments: W1 fp32 stream (b128) -> regs -> bf16 -> double-buffered LDS
//   8 WMMAs / wave / iteration; grouped ds_load -> sched_barrier -> wmma
// ---------------------------------------------------------------------------
__global__ __launch_bounds__(256) void moe_ffn_kernel(
    const unsigned* __restrict__ xbf, const float* __restrict__ W1,
    const float* __restrict__ b1,     const float* __restrict__ W2,
    float* __restrict__ ypart)
{
    __shared__ unsigned bs[2][HT * (KT / 2)];   // 2 x (128 rows x 16 bf16-pairs) = 16 KB
    __shared__ float    hs[NB * HT];            // silu outputs (64 KB)
    __shared__ float    w2s[OO * HT];           // W2 slice (5 KB)
    __shared__ float    b1s[HT];

    const int e     = blockIdx.y;
    const int h0    = blockIdx.x * HT;
    const int t     = threadIdx.x;
    const int lane  = t & 31;
    const int wv    = t >> 5;        // wave id 0..7
    const int mtile = wv * 16;       // this wave's 16-token row block

    if (t < HT) b1s[t] = b1[e * HH + h0 + t];
    for (int i = t; i < OO * HT; i += 256) {
        int o = i / HT, nn = i % HT;
        w2s[i] = W2[((size_t)e * OO + o) * HH + h0 + nn];
    }

    FragC acc[NSUB];
#pragma unroll
    for (int s = 0; s < NSUB; ++s) acc[s].v = (v8f){};

    const float* w1base = W1 + (size_t)e * HH * DD + (size_t)h0 * DD;
    const int sr = t >> 3;           // staged row base (q>>3 for q=t)
    const int sc = t & 7;            // staged float4-column within row

    // per-thread register staging of one W1 tile: 4x float4 (rows sr, sr+32, ...)
    float4 rg[4];
#pragma unroll
    for (int i = 0; i < 4; ++i)
        rg[i] = *(const float4*)(w1base + (size_t)(i * 32 + sr) * DD + 0 + sc * 4);

    // A-fragment addressing (16-bit A 16x32, contiguous k = 16*(lane/16) + s, same order as B)
    const int arow = mtile + (lane & 15);
    const int alo  = (lane < 16) ? 0 : 8;
    const unsigned* xrow = xbf + (size_t)arow * (DD / 2);

    int buf = 0;
    for (int k0 = 0; k0 < DD; k0 += KT) {
        // commit staged regs (bf16 pack) into LDS tile: 2 pairs per float4
#pragma unroll
        for (int i = 0; i < 4; ++i) {
            unsigned* dst = &bs[buf][(i * 32 + sr) * 16 + sc * 2];
            uint2 pp; pp.x = pack2(rg[i].x, rg[i].y); pp.y = pack2(rg[i].z, rg[i].w);
            *(uint2*)dst = pp;                       // ds_store_b64
        }
        __syncthreads();

        // start streaming the next W1 tile into registers (hidden under WMMA)
        if (k0 + KT < DD) {
#pragma unroll
            for (int i = 0; i < 4; ++i)
                rg[i] = *(const float4*)(w1base + (size_t)(i * 32 + sr) * DD + (k0 + KT) + sc * 4);
        }
        // prefetch two tiles ahead on the HBM stream
        if (k0 + 2 * KT < DD) {
#pragma unroll
            for (int i = 0; i < 4; ++i)
                __builtin_prefetch(w1base + (size_t)(i * 32 + sr) * DD + (k0 + 2 * KT) + sc * 4, 0, 1);
        }

        // A fragment: two b128 loads straight from L2-resident bf16 x
        FragBF a;
        {
            const uint4 lo4 = *(const uint4*)(xrow + (k0 >> 1) + alo);
            const uint4 hi4 = *(const uint4*)(xrow + (k0 >> 1) + 4 + alo);
            a.u[0] = lo4.x; a.u[1] = lo4.y; a.u[2] = lo4.z; a.u[3] = lo4.w;
            a.u[4] = hi4.x; a.u[5] = hi4.y; a.u[6] = hi4.z; a.u[7] = hi4.w;
        }

        // B fragments: two groups of 4; fence so ds_loads batch before WMMAs
        const int kb = (lane < 16) ? 0 : 8;
#pragma unroll
        for (int g = 0; g < 2; ++g) {
            FragBF bq[4];
#pragma unroll
            for (int s = 0; s < 4; ++s) {
                int col = (g * 4 + s) * 16 + (lane & 15);
#pragma unroll
                for (int v = 0; v < 8; ++v) bq[s].u[v] = bs[buf][col * 16 + kb + v];
            }
            __builtin_amdgcn_sched_barrier(0);
#pragma unroll
            for (int s = 0; s < 4; ++s)
                acc[g * 4 + s].v = __builtin_amdgcn_wmma_f32_16x16x32_bf16(
                    false, a.v, false, bq[s].v, (short)0, acc[g * 4 + s].v, false, false);
            __builtin_amdgcn_sched_barrier(0);
        }
        buf ^= 1;
    }

    // epilogue: bias + silu, scatter to LDS per C layout (ISA 7.12.2)
#pragma unroll
    for (int s = 0; s < NSUB; ++s) {
#pragma unroll
        for (int j = 0; j < 8; ++j) {
            int m = mtile + j + ((lane < 16) ? 0 : 8);
            int n = s * 16 + (lane & 15);
            float v  = acc[s].f[j] + b1s[n];
            float sg = __builtin_amdgcn_rcpf(1.0f + __expf(-v));
            hs[m * HT + n] = v * sg;
        }
    }
    __syncthreads();

    // FFN2 slice: 128 tokens x 10 outputs, HT-term dots, accumulate globally
#pragma unroll
    for (int i = 0; i < 5; ++i) {
        int task = i * 256 + t;                  // 0..1279
        int tok = task / OO, o = task % OO;
        float a2 = 0.0f;
#pragma unroll 8
        for (int nn = 0; nn < HT; ++nn)
            a2 += hs[tok * HT + nn] * w2s[o * HT + nn];
        atomicAdd(&ypart[(tok * NE + e) * OO + o], a2);
    }
}

// ---------------------------------------------------------------------------
// Kernel 3: out[b,o] = sum_e w[b,e] * (ypart[b,e,o] + b2[e,o])
// ---------------------------------------------------------------------------
__global__ __launch_bounds__(256) void moe_final_kernel(
    const float* __restrict__ wbuf, const float* __restrict__ ypart,
    const float* __restrict__ b2, float* __restrict__ out)
{
    int i = blockIdx.x * 256 + threadIdx.x;
    if (i >= NB * OO) return;
    int b = i / OO, o = i % OO;
    float acc = 0.0f;
#pragma unroll
    for (int e = 0; e < NE; ++e)
        acc += wbuf[b * NE + e] * (ypart[(b * NE + e) * OO + o] + b2[e * OO + o]);
    out[i] = acc;
}

extern "C" void kernel_launch(void* const* d_in, const int* in_sizes, int n_in,
                              void* d_out, int out_size, void* d_ws, size_t ws_size,
                              hipStream_t stream) {
    const float* E      = (const float*)d_in[0];
    const float* gate_w = (const float*)d_in[1];
    const float* gate_b = (const float*)d_in[2];
    const float* W1     = (const float*)d_in[3];
    const float* b1     = (const float*)d_in[4];
    const float* W2     = (const float*)d_in[5];
    const float* b2     = (const float*)d_in[6];
    float* out  = (float*)d_out;

    float*    wbuf  = (float*)d_ws;                 // 128*8 gate weights
    float*    ypart = wbuf + NB * NE;               // 128*8*10 partial y
    unsigned* xbf   = (unsigned*)(ypart + NB * NE * OO);  // 128 x 1024 bf16-pairs (512 KB)

    moe_gate_kernel<<<NB, 256, 0, stream>>>(E, gate_w, gate_b, wbuf, ypart, xbf);
    moe_ffn_kernel<<<dim3(HH / HT, NE), 256, 0, stream>>>(xbf, W1, b1, W2, ypart);
    moe_final_kernel<<<(NB * OO + 255) / 256, 256, 0, stream>>>(wbuf, ypart, b2, out);
}
